// mLSTMBackend_28226525070271
// MI455X (gfx1250) — matrix operand order repaired and checked by reading, not compiled, not run on hardware
//
#include <hip/hip_runtime.h>

// ---------------------------------------------------------------------------
// mLSTM forward for MI455X (gfx1250, wave32, WMMA).
// Parallel (flash-attention-style) exact reformulation of the scan:
//   m_t = F_t + M_t,  F_t = cumsum(logsigmoid(f)),  M_t = prefixmax(i_s - F_s)
//   P[t,s] = (q_t.k_s * DH^-0.5) * exp((F_t - m_t) + (i_s - F_s)),  s <= t
//   h_t = (P V)_t / (max(|rowsum P|_t, exp(-m_t)) + eps)
// GEMMs run on v_wmma_f32_16x16x32_bf16; gating/exp/normalize in f32 VALU.
// Gate prefix scans are a two-level parallel scan (one block per head).
// ---------------------------------------------------------------------------

#define S_LEN   2048
#define D_HEAD  128
#define N_HEADS 8          // B*NH = 2*4
#define BQ      64         // query rows per workgroup (4 waves x 16 rows)
#define CK      32         // kv chunk = WMMA K for bf16
#define NWAVE   4
#define KPAD    136        // 128 + 8 bf16 pad (row stride 272B = 16B aligned)
#define VPAD    40         // 32 + 8 pad   (row stride 80B = 16B aligned)
#define PPAD    36         // 32 + 4 pad
#define SCAN_T  256        // threads per gate-scan block
#define SEG     (S_LEN / SCAN_T)   // 8 elements per thread

typedef __attribute__((ext_vector_type(16))) __bf16 v16bf;
typedef __attribute__((ext_vector_type(8)))  float  v8f;

// ---------------------- gate prefix scans (parallel) -----------------------
// Per head: F_t = cumsum(logsigmoid(f_t)), M_t = prefixmax(i_t - F_t),
// outputs rowA = F-m = -M, colB = i-F, mexp = exp(-m) with m = F + M.
__global__ __launch_bounds__(SCAN_T, 1)
void mlstm_gate_scan(const float* __restrict__ ig,
                     const float* __restrict__ fg,
                     float* __restrict__ rowA,
                     float* __restrict__ colB,
                     float* __restrict__ mexp) {
    __shared__ float sSum[SCAN_T];
    __shared__ float sMax[SCAN_T];

    const int h   = blockIdx.x;
    const int tid = threadIdx.x;
    const float* ip = ig + (size_t)h * S_LEN + tid * SEG;
    const float* fp = fg + (size_t)h * S_LEN + tid * SEG;

    // ---- phase 1: per-thread segment scan (registers)
    float fl[SEG], cl[SEG];          // logsigmoid(f), local c = i - Flocal
    float Floc = 0.0f;
    float Mloc = -3.0e38f;
#pragma unroll
    for (int e = 0; e < SEG; ++e) {
        const float fv = fp[e];
        fl[e] = (fv >= 0.0f) ? -log1pf(__expf(-fv))
                             : (fv - log1pf(__expf(fv)));
        Floc += fl[e];
        cl[e] = ip[e] - Floc;
        Mloc  = fmaxf(Mloc, cl[e]);
    }

    // ---- phase 2a: exclusive prefix-sum of segment sums -> Fstart
    sSum[tid] = Floc;
    __syncthreads();
#pragma unroll
    for (int off = 1; off < SCAN_T; off <<= 1) {
        const float x = (tid >= off) ? sSum[tid - off] : 0.0f;
        __syncthreads();
        sSum[tid] += x;
        __syncthreads();
    }
    const float Fstart = (tid > 0) ? sSum[tid - 1] : 0.0f;

    // ---- phase 2b: exclusive prefix-max of (Mloc - Fstart) -> Mexcl
    sMax[tid] = Mloc - Fstart;
    __syncthreads();
#pragma unroll
    for (int off = 1; off < SCAN_T; off <<= 1) {
        const float x = (tid >= off) ? sMax[tid - off] : -3.0e38f;
        __syncthreads();
        sMax[tid] = fmaxf(sMax[tid], x);
        __syncthreads();
    }
    const float Mexcl = (tid > 0) ? sMax[tid - 1] : -3.0e38f;

    // ---- phase 3: emit per-element results
    float F = Fstart;
    float Mrun = Mexcl;
    float* rA = rowA + (size_t)h * S_LEN + tid * SEG;
    float* cB = colB + (size_t)h * S_LEN + tid * SEG;
    float* mx = mexp + (size_t)h * S_LEN + tid * SEG;
#pragma unroll
    for (int e = 0; e < SEG; ++e) {
        F += fl[e];
        const float cglob = cl[e] - Fstart;     // i_t - F_t (global)
        Mrun = fmaxf(Mrun, cglob);
        rA[e] = -Mrun;                          // F - m
        cB[e] = cglob;                          // i - F
        mx[e] = __expf(-(F + Mrun));            // exp(-m)
    }
}

// ------------------------- staged KV -> LDS (bf16) -------------------------
__device__ __forceinline__ void stage_kv(const float* __restrict__ kh,
                                         const float* __restrict__ vh,
                                         int s0, __bf16* sKb, __bf16* sVb,
                                         int tid) {
    const int e   = tid * 32;          // 128 threads * 32 = 32*128 elements
    const int row = e >> 7;            // kv row in chunk (0..31)
    const int col = e & 127;           // dh start
    const float* ks = kh + (size_t)(s0 + row) * D_HEAD + col;
    const float* vs = vh + (size_t)(s0 + row) * D_HEAD + col;
#pragma unroll
    for (int j = 0; j < 8; ++j) {
        const float4 kf = *(const float4*)(ks + 4 * j);
        const float4 vf = *(const float4*)(vs + 4 * j);
        const int c = col + 4 * j;
        __bf16* kd = sKb + row * KPAD + c;             // K: row-major [kv][dh]
        kd[0] = (__bf16)kf.x; kd[1] = (__bf16)kf.y;
        kd[2] = (__bf16)kf.z; kd[3] = (__bf16)kf.w;
        sVb[(c + 0) * VPAD + row] = (__bf16)vf.x;      // V: transposed [dh][kv]
        sVb[(c + 1) * VPAD + row] = (__bf16)vf.y;
        sVb[(c + 2) * VPAD + row] = (__bf16)vf.z;
        sVb[(c + 3) * VPAD + row] = (__bf16)vf.w;
    }
}

// ------------------------------ main kernel --------------------------------
__global__ __launch_bounds__(128, 1)
void mlstm_fwd(const float* __restrict__ Q, const float* __restrict__ K,
               const float* __restrict__ V, const float* __restrict__ rowA,
               const float* __restrict__ colB, const float* __restrict__ mexp,
               float* __restrict__ O) {
    __shared__ __align__(16) __bf16 sK[2][CK * KPAD];
    __shared__ __align__(16) __bf16 sV[2][D_HEAD * VPAD];
    __shared__ __align__(16) __bf16 sP[NWAVE][16 * PPAD];

    const int h   = blockIdx.y;
    const int qt  = blockIdx.x;
    const int tid = threadIdx.x;
    const int wv  = tid >> 5;
    const int ln  = tid & 31;
    const int hf  = ln >> 4;     // lane half: D rows m = v + 8*hf, A/B K offset
    const int lr  = ln & 15;
    const int r0  = qt * BQ + wv * 16;

    const float* qh = Q + (size_t)h * S_LEN * D_HEAD;
    const float* kh = K + (size_t)h * S_LEN * D_HEAD;
    const float* vh = V + (size_t)h * S_LEN * D_HEAD;

    // ---- Q tile (16 rows) into 4 bf16 A-fragments, pre-scaled by DH^-0.5
    const float qs = 0.08838834764831845f;
    v16bf aQ[4];
    {
        const float* qr = qh + (size_t)(r0 + lr) * D_HEAD;
#pragma unroll
        for (int f = 0; f < 4; ++f) {
#pragma unroll
            for (int d = 0; d < 8; ++d) {
                const int kk = f * 32 + hf * 16 + d * 2;
                const float2 x = *(const float2*)(qr + kk);
                aQ[f][2 * d]     = (__bf16)(x.x * qs);
                aQ[f][2 * d + 1] = (__bf16)(x.y * qs);
            }
        }
    }
    float rA[8];
#pragma unroll
    for (int vv = 0; vv < 8; ++vv)
        rA[vv] = rowA[h * S_LEN + r0 + hf * 8 + vv];

    v8f acc[8];
#pragma unroll
    for (int j = 0; j < 8; ++j) acc[j] = (v8f){};
    float qn[8] = {0.f, 0.f, 0.f, 0.f, 0.f, 0.f, 0.f, 0.f};

    const int nch = (qt + 1) * (BQ / CK);   // causal chunk count

    stage_kv(kh, vh, 0, sK[0], sV[0], tid);
    __syncthreads();

    for (int c = 0; c < nch; ++c) {
        const int buf = c & 1;
        const int s0  = c * CK;
        if (c + 1 < nch)
            stage_kv(kh, vh, s0 + CK, sK[buf ^ 1], sV[buf ^ 1], tid);

        if (s0 <= r0 + 15) {                // this wave has unmasked work
            // ---- scores: two 16x16 tiles, contraction over DH=128
            v8f sc[2];
#pragma unroll
            for (int nt = 0; nt < 2; ++nt) {
                v8f a = (v8f){};
#pragma unroll
                for (int f = 0; f < 4; ++f) {
                    const v16bf bk = *(const v16bf*)
                        (sK[buf] + (nt * 16 + lr) * KPAD + f * 32 + hf * 16);
                    a = __builtin_amdgcn_wmma_f32_16x16x32_bf16(
                            false, aQ[f], false, bk, (short)0, a, false, false);
                }
                sc[nt] = a;
            }
            // ---- gate + causal mask + row-sum + spill P (bf16) to LDS
            const bool diag = (s0 + CK - 1 > r0);
#pragma unroll
            for (int nt = 0; nt < 2; ++nt) {
                const int   scol = s0 + nt * 16 + lr;
                const float cB   = colB[h * S_LEN + scol];
#pragma unroll
                for (int vv = 0; vv < 8; ++vv) {
                    float p = sc[nt][vv] * __expf(rA[vv] + cB);
                    if (diag && scol > r0 + hf * 8 + vv) p = 0.0f;
                    qn[vv] += p;
                    sP[wv][(hf * 8 + vv) * PPAD + nt * 16 + lr] = (__bf16)p;
                }
            }
            // ---- reload P as A-fragment (layout shuffle via LDS, same wave)
            v16bf aP;
            {
                const __bf16* pp = sP[wv] + lr * PPAD;
#pragma unroll
                for (int d = 0; d < 8; ++d) {
                    const int kk = hf * 16 + d * 2;
                    aP[2 * d]     = pp[kk];
                    aP[2 * d + 1] = pp[kk + 1];
                }
            }
            // ---- P @ V : 8 output column tiles of DH
#pragma unroll
            for (int j = 0; j < 8; ++j) {
                const v16bf bv = *(const v16bf*)
                    (sV[buf] + (j * 16 + lr) * VPAD + hf * 16);
                acc[j] = __builtin_amdgcn_wmma_f32_16x16x32_bf16(
                             false, aP, false, bv, (short)0, acc[j], false, false);
            }
        }
        __syncthreads();
    }

    // ---- reduce qn across the 16 lanes holding the same rows
#pragma unroll
    for (int m = 1; m < 16; m <<= 1)
#pragma unroll
        for (int vv = 0; vv < 8; ++vv)
            qn[vv] += __shfl_xor(qn[vv], m, 32);

    float rd[8];
#pragma unroll
    for (int vv = 0; vv < 8; ++vv) {
        const int t = r0 + hf * 8 + vv;
        const float dn = fmaxf(fabsf(qn[vv]), mexp[h * S_LEN + t]) + 1e-6f;
        rd[vv] = 1.0f / dn;
    }
    float* oh = O + (size_t)h * S_LEN * D_HEAD;
#pragma unroll
    for (int j = 0; j < 8; ++j)
#pragma unroll
        for (int vv = 0; vv < 8; ++vv)
            oh[(size_t)(r0 + hf * 8 + vv) * D_HEAD + j * 16 + lr] =
                acc[j][vv] * rd[vv];
}

// ------------------------------- launcher ----------------------------------
extern "C" void kernel_launch(void* const* d_in, const int* in_sizes, int n_in,
                              void* d_out, int out_size, void* d_ws,
                              size_t ws_size, hipStream_t stream) {
    (void)in_sizes; (void)n_in; (void)out_size; (void)ws_size;
    const float* q  = (const float*)d_in[0];
    const float* k  = (const float*)d_in[1];
    const float* v  = (const float*)d_in[2];
    const float* ig = (const float*)d_in[3];
    const float* fg = (const float*)d_in[4];

    float* ws   = (float*)d_ws;
    float* rowA = ws;
    float* colB = ws + (size_t)N_HEADS * S_LEN;
    float* mexp = ws + (size_t)2 * N_HEADS * S_LEN;

    mlstm_gate_scan<<<N_HEADS, SCAN_T, 0, stream>>>(ig, fg, rowA, colB, mexp);

    dim3 grid(S_LEN / BQ, N_HEADS);
    mlstm_fwd<<<grid, 128, 0, stream>>>(q, k, v, rowA, colB, mexp,
                                        (float*)d_out);
}
